// GraphAttentionLayer_80745385164828
// MI455X (gfx1250) — matrix operand order repaired and checked by reading, not compiled, not run on hardware
//
#include <hip/hip_runtime.h>
#include <hip/hip_bf16.h>
#include <math.h>

#define B_    8
#define N_    2048
#define FIN   256
#define FOUT  128

typedef float  v2f   __attribute__((ext_vector_type(2)));
typedef float  v8f   __attribute__((ext_vector_type(8)));
typedef __bf16 v16bf __attribute__((ext_vector_type(16)));
typedef int    v4i   __attribute__((ext_vector_type(4)));
typedef __attribute__((address_space(1))) v4i gv4i;   // global int4
typedef __attribute__((address_space(3))) v4i lv4i;   // LDS int4

#if defined(__has_builtin)
#if __has_builtin(__builtin_amdgcn_global_load_async_to_lds_b128) && \
    __has_builtin(__builtin_amdgcn_s_wait_asynccnt)
#define GAT_ASYNC_LDS 1
#endif
#endif

// ---------------------------------------------------------------------------
// Kernel 1: Wh = x @ W via V_WMMA_F32_16X16X4_F32 (full fp32 precision).
// Each wave: one 16-row strip x all 128 cols (8 C tiles). W staged into LDS in
// two 128-row K-phases (64KB) with GLOBAL_LOAD_ASYNC_TO_LDS when available.
// Epilogue: f1/f2 row dots (shfl reduction) and Wh^T written in bf16 for the
// attention GEMM's B-operand layout.
// ---------------------------------------------------------------------------
__global__ __launch_bounds__(256)
void gat_gemm1_kernel(const float* __restrict__ x, const float* __restrict__ W,
                      const float* __restrict__ avec,
                      __bf16* __restrict__ whT,
                      float* __restrict__ f1, float* __restrict__ f2)
{
    __shared__ float sW[128 * FOUT];                 // 64 KB: one K-half of W
    const int wave = threadIdx.x >> 5;
    const int lane = threadIdx.x & 31;
    const int half = lane >> 4;                      // 0: K+0/1 ; 1: K+2/3
    const int l16  = lane & 15;
    const int b    = blockIdx.x >> 4;                // 16 blocks per batch
    const int i0   = ((blockIdx.x & 15) * 8 + wave) * 16;
    const float* xrow = x + (size_t)(b * N_ + i0 + l16) * FIN;

    v8f c[8] = {};

    for (int phase = 0; phase < 2; ++phase) {
        const float* wsrc = W + phase * 128 * FOUT;
#ifdef GAT_ASYNC_LDS
        for (int idx = threadIdx.x; idx < (128 * FOUT) / 4; idx += 256) {
            __builtin_amdgcn_global_load_async_to_lds_b128(
                (gv4i*)(wsrc + idx * 4),
                (lv4i*)(sW + idx * 4),
                0, 0);
        }
        __builtin_amdgcn_s_wait_asynccnt(0);
#else
        for (int idx = threadIdx.x; idx < (128 * FOUT) / 4; idx += 256)
            ((float4*)sW)[idx] = ((const float4*)wsrc)[idx];
#endif
        __syncthreads();

        for (int k = 0; k < 128; k += 4) {
            // A frag: lane half selects K pair (ISA 32-bit A 16x4 layout)
            const float* ap = xrow + phase * 128 + k + 2 * half;
            v2f a; a.x = ap[0]; a.y = ap[1];
            #pragma unroll
            for (int t = 0; t < 8; ++t) {
                const int col = t * 16 + l16;
                v2f bb;
                bb.x = sW[(k + 2 * half + 0) * FOUT + col];
                bb.y = sW[(k + 2 * half + 1) * FOUT + col];
                c[t] = __builtin_amdgcn_wmma_f32_16x16x4_f32(
                           false, a, false, bb, (short)0, c[t], false, false);
            }
        }
        __syncthreads();
    }

    // f1/f2 from C frags + Wh^T (bf16) stores.
    float a1v[8], a2v[8];
    #pragma unroll
    for (int t = 0; t < 8; ++t) {
        a1v[t] = avec[t * 16 + l16];
        a2v[t] = avec[FOUT + t * 16 + l16];
    }
    #pragma unroll
    for (int v = 0; v < 8; ++v) {
        const int row = i0 + v + 8 * half;           // C layout: M = v (+8 for hi half)
        float s1 = 0.f, s2 = 0.f;
        #pragma unroll
        for (int t = 0; t < 8; ++t) { s1 += c[t][v] * a1v[t]; s2 += c[t][v] * a2v[t]; }
        #pragma unroll
        for (int m = 1; m < 16; m <<= 1) {           // reduce within 16-lane half
            s1 += __shfl_xor(s1, m, 32);
            s2 += __shfl_xor(s2, m, 32);
        }
        if (l16 == 0) { f1[b * N_ + row] = s1; f2[b * N_ + row] = s2; }
        #pragma unroll
        for (int t = 0; t < 8; ++t)
            whT[(size_t)(b * FOUT + t * 16 + l16) * N_ + row] = (__bf16)c[t][v];
    }
}

// ---------------------------------------------------------------------------
// Kernel 2: single pass over adj (the only full 128MB read). Produces
//  (a) per-row masked max of f2 (lrelu is monotone -> softmax row max), and
//  (b) adj compressed to a bitmask in natural j-order via wave32 ballot,
//      so the attention kernel reads 1 bit/edge instead of 4 bytes (32x).
// One wave per (b,i) row; lane L owns j = it*32 + L.
// ---------------------------------------------------------------------------
__global__ __launch_bounds__(256)
void gat_rowmax_kernel(const int* __restrict__ adj, const float* __restrict__ f1,
                       const float* __restrict__ f2, float* __restrict__ emax,
                       unsigned* __restrict__ amask)
{
    const int wave = threadIdx.x >> 5;
    const int lane = threadIdx.x & 31;
    const int r = blockIdx.x * 8 + wave;             // 16384 rows total
    const int b = r >> 11;
    const int* arow = adj + (size_t)r * N_;
    const float* f2b = f2 + b * N_;
    unsigned* mrow = amask + (size_t)r * (N_ / 32);

    float m = -3.0e38f;
    for (int it = 0; it < N_ / 32; ++it) {
        const int j = it * 32 + lane;
        const int   av = arow[j];
        const float fv = f2b[j];
        const unsigned bal = (unsigned)__ballot(av > 0);   // bit L <-> j = it*32+L
        if (av > 0) m = fmaxf(m, fv);
        if (lane == 0) mrow[it] = bal;
    }
    #pragma unroll
    for (int s = 1; s < 32; s <<= 1) m = fmaxf(m, __shfl_xor(m, s, 32));
    if (lane == 0) {
        float e = f1[r] + m;
        e = (e >= 0.f) ? e : 2.f * e;                // leaky, alpha = 2
        if (m <= -1.0e30f) e = 0.f;                  // isolated-node guard
        emax[r] = e;
    }
}

// ---------------------------------------------------------------------------
// Kernel 3: h' = softmax_row(mask(lrelu(f1+f2))) @ Wh, fused, bf16 WMMA.
// One wave per 16-row i-tile; K(=j) stepped by 32 per v_wmma_f32_16x16x32_bf16.
// A frag (probs) built in-register in contiguous k order (k = 16*half + slot) using the bitmask;
// B frag is one contiguous 32B load of Wh^T. Unnormalized accumulate + scalar
// row sum, then one normalize + ELU epilogue.
// ---------------------------------------------------------------------------
#define GAT_P(slot, bit, fv)                                                   \
    do { float e_ = f1r + (fv); e_ = (e_ >= 0.f) ? e_ : 2.f * e_;              \
         float p_ = (bit) ? __expf(e_ - emr) : 0.f;                            \
         ssum += p_; pa[slot] = (__bf16)p_; } while (0)

__global__ __launch_bounds__(128)
void gat_attn_kernel(const unsigned* __restrict__ amask,
                     const __bf16* __restrict__ whT,
                     const float* __restrict__ f1, const float* __restrict__ f2,
                     const float* __restrict__ emax, float* __restrict__ out)
{
    const int wave = threadIdx.x >> 5;
    const int lane = threadIdx.x & 31;
    const int half = lane >> 4;
    const int l16  = lane & 15;
    const int tile = blockIdx.x * 4 + wave;          // 1024 i-tiles total
    const int b    = tile >> 7;
    const int i0   = (tile & 127) * 16;
    const int row  = i0 + l16;

    const float f1r = f1[b * N_ + row];
    const float emr = emax[b * N_ + row];
    const unsigned* mrow  = amask + (size_t)(b * N_ + row) * (N_ / 32);
    const float*    f2b   = f2 + b * N_;
    const __bf16*   wbase = whT + (size_t)b * FOUT * N_;

    v8f c[8] = {};
    float ssum = 0.f;

    for (int j0 = 0; j0 < N_; j0 += 32) {
        const int base0 = j0 + 16 * half;            // A-frag K slots 0..7
        const int base1 = j0 + 16 * half + 8;        // A-frag K slots 8..15
        const unsigned w     = mrow[j0 >> 5];
        const unsigned bits0 = (w >> (16 * half)) & 0xffu;
        const unsigned bits1 = (w >> (16 * half + 8)) & 0xffu;
        float4 g0 = *(const float4*)(f2b + base0);
        float4 g1 = *(const float4*)(f2b + base0 + 4);
        float4 g2 = *(const float4*)(f2b + base1);
        float4 g3 = *(const float4*)(f2b + base1 + 4);

        v16bf pa;
        GAT_P( 0, bits0 & 0x01u, g0.x); GAT_P( 1, bits0 & 0x02u, g0.y);
        GAT_P( 2, bits0 & 0x04u, g0.z); GAT_P( 3, bits0 & 0x08u, g0.w);
        GAT_P( 4, bits0 & 0x10u, g1.x); GAT_P( 5, bits0 & 0x20u, g1.y);
        GAT_P( 6, bits0 & 0x40u, g1.z); GAT_P( 7, bits0 & 0x80u, g1.w);
        GAT_P( 8, bits1 & 0x01u, g2.x); GAT_P( 9, bits1 & 0x02u, g2.y);
        GAT_P(10, bits1 & 0x04u, g2.z); GAT_P(11, bits1 & 0x08u, g2.w);
        GAT_P(12, bits1 & 0x10u, g3.x); GAT_P(13, bits1 & 0x20u, g3.y);
        GAT_P(14, bits1 & 0x40u, g3.z); GAT_P(15, bits1 & 0x80u, g3.w);

        #pragma unroll
        for (int t = 0; t < 8; ++t) {
            // B frag: column (t*16+l16) of Wh == row of Wh^T, K = j0+16*half..+15
            const __bf16* bp = wbase + (size_t)(t * 16 + l16) * N_ + j0 + 16 * half;
            v16bf bb = *(const v16bf*)bp;
            c[t] = __builtin_amdgcn_wmma_f32_16x16x32_bf16(
                       false, pa, false, bb, (short)0, c[t], false, false);
        }
    }

    // Lanes L and L+16 jointly cover all j for row l16.
    ssum += __shfl_xor(ssum, 16, 32);
    const float inv = 1.0f / fmaxf(ssum, 1e-30f);

    #pragma unroll
    for (int v = 0; v < 8; ++v) {
        const float iv   = __shfl(inv, v + 8 * half, 32);   // bcast row sum
        const int   orow = i0 + v + 8 * half;
        float* op = out + (size_t)(b * N_ + orow) * FOUT;
        #pragma unroll
        for (int t = 0; t < 8; ++t) {
            float val = c[t][v] * iv;
            val = (val > 0.f) ? val : expm1f(val);          // ELU
            op[t * 16 + l16] = val;
        }
    }
}

// ---------------------------------------------------------------------------
extern "C" void kernel_launch(void* const* d_in, const int* in_sizes, int n_in,
                              void* d_out, int out_size, void* d_ws, size_t ws_size,
                              hipStream_t stream)
{
    (void)in_sizes; (void)n_in; (void)out_size; (void)ws_size;
    const float* x   = (const float*)d_in[0];
    const int*   adj = (const int*)d_in[1];
    const float* W   = (const float*)d_in[2];
    const float* a   = (const float*)d_in[3];
    float*       out = (float*)d_out;

    char* ws = (char*)d_ws;
    __bf16*   whT   = (__bf16*)ws;                                 // 4 MB (B*F*N bf16)
    unsigned* amask = (unsigned*)(ws + (size_t)4 * 1024 * 1024);   // 4 MB bitmask
    float*    f1    = (float*)(ws + (size_t)8 * 1024 * 1024);      // 64 KB
    float*    f2    = f1 + B_ * N_;                                // 64 KB
    float*    emax  = f2 + B_ * N_;                                // 64 KB

    gat_gemm1_kernel <<<128,  256, 0, stream>>>(x, W, a, whT, f1, f2);
    gat_rowmax_kernel<<<2048, 256, 0, stream>>>(adj, f1, f2, emax, amask);
    gat_attn_kernel  <<<256,  128, 0, stream>>>(amask, whT, f1, f2, emax, out);
}
